// SceneBoundaryTemporalEmbedding_33887291965510
// MI455X (gfx1250) — matrix operand order repaired and checked by reading, not compiled, not run on hardware
//
#include <hip/hip_runtime.h>
#include <math.h>

typedef __attribute__((ext_vector_type(16))) _Float16 v16h;
typedef __attribute__((ext_vector_type(8)))  _Float16 v8h;
typedef __attribute__((ext_vector_type(8)))  float    v8f;

#define Bz     32
#define Kz     2048
#define Dz     768
#define Hz     1024
#define HALFz  512
#define MAXLEN 4500
#define MPAD   4512               // MAXLEN padded to multiple of 16
#define Mrows  (Bz * Kz)          // 65536 output rows

__device__ __forceinline__ float gelu_exact(float x) {
    return 0.5f * x * (1.0f + erff(x * 0.70710678118654752f));
}

// ---------------------------------------------------------------------------
// Split Wm (1024x1024) into f16 halves: WmAh[o,h]=Wm[o,h], WmBh[o,g]=Wm[o,512+g]
// ---------------------------------------------------------------------------
__global__ void k_cvt_wm(const float* __restrict__ Wm,
                         _Float16* __restrict__ WmAh, _Float16* __restrict__ WmBh) {
    int t = blockIdx.x * blockDim.x + threadIdx.x;
    if (t >= Hz * Hz) return;
    int o = t >> 10, h = t & (Hz - 1);
    _Float16 v = (_Float16)Wm[t];
    if (h < HALFz) WmAh[o * HALFz + h] = v;
    else           WmBh[o * HALFz + (h - HALFz)] = v;
}

// W2t[h,g] = W2[g,h]  (512x512 f16 transpose)
__global__ void k_t_w2(const float* __restrict__ W2, _Float16* __restrict__ W2t) {
    int t = blockIdx.x * blockDim.x + threadIdx.x;
    if (t >= HALFz * HALFz) return;
    int g = t >> 9, h = t & (HALFz - 1);
    W2t[h * HALFz + g] = (_Float16)W2[t];
}

// peh: abs_pe (4500x512) -> f16, zero-padded to 4512 rows
__global__ void k_cvt_pe(const float* __restrict__ pe, _Float16* __restrict__ peh) {
    int t = blockIdx.x * blockDim.x + threadIdx.x;
    if (t >= MPAD * HALFz) return;
    int p = t >> 9;
    peh[t] = (p < MAXLEN) ? (_Float16)pe[t] : (_Float16)0.f;
}

// bc[o] = bm[o] + sum_g b2[g] * Wm[o, 512+g]
__global__ void k_bconst(const float* __restrict__ bm, const float* __restrict__ b2,
                         const float* __restrict__ Wm, float* __restrict__ bc) {
    int o = blockIdx.x * blockDim.x + threadIdx.x;
    if (o >= Hz) return;
    const float* w = Wm + (size_t)o * Hz + HALFz;
    float acc = bm[o];
    for (int g = 0; g < HALFz; ++g) acc += b2[g] * w[g];
    bc[o] = acc;
}

// ---------------------------------------------------------------------------
// Generic NT WMMA GEMM: C[m,n] = sum_k A[m,k]*Bm[n,k], f16 in, f32 or f16 out.
// Block 128 threads (4 waves), tile 16 x 64. grid = (M/16, N/64). K % 32 == 0.
// ---------------------------------------------------------------------------
template <bool OUT_HALF>
__global__ void __launch_bounds__(128)
k_gemm_nt(const _Float16* __restrict__ A, const _Float16* __restrict__ Bm,
          void* __restrict__ Cv, int N, int K) {
    int mtile = blockIdx.x, ntile = blockIdx.y;
    int tid = threadIdx.x, wave = tid >> 5, lane = tid & 31;
    int nsub = lane & 15;
    int col0 = ntile * 64 + wave * 16;
    int akb = (lane < 16) ? 0 : 8;     // A: 16x32 f16 fragment layout
    int bkb = (lane < 16) ? 0 : 8;     // B: 32x16 f16 fragment layout (same k order as A)
    const _Float16* arow = A + (size_t)(mtile * 16 + (lane & 15)) * K;
    const _Float16* brow = Bm + (size_t)(col0 + nsub) * K;
    v8f c = {};
    for (int kk = 0; kk < K / 32; ++kk) {
        const _Float16* ap = arow + kk * 32 + akb;
        v8h alo = *(const v8h*)(ap);
        v8h ahi = *(const v8h*)(ap + 16);
        v16h a = __builtin_shufflevector(alo, ahi,
                 0,1,2,3,4,5,6,7,8,9,10,11,12,13,14,15);
        const _Float16* bp = brow + kk * 32 + bkb;
        v8h blo = *(const v8h*)(bp);
        v8h bhi = *(const v8h*)(bp + 16);
        v16h bf = __builtin_shufflevector(blo, bhi,
                 0,1,2,3,4,5,6,7,8,9,10,11,12,13,14,15);
        c = __builtin_amdgcn_wmma_f32_16x16x32_f16(
                false, a, false, bf, (short)0, c, false, false);
    }
    int mplus = (lane < 16) ? 0 : 8;
    int o = col0 + nsub;
    #pragma unroll
    for (int r = 0; r < 8; ++r) {
        size_t m = (size_t)(mtile * 16 + r + mplus);
        if (OUT_HALF) ((_Float16*)Cv)[m * N + o] = (_Float16)c[r];
        else          ((float*)Cv)[m * N + o] = c[r];
    }
}

// ---------------------------------------------------------------------------
// Neighbor cosine sims -> boundary mask. One wave per 64-pair strip; previous
// row kept in registers so each fe row is read ~once (201 MB total).
// grid.x = B*4, block 256 (8 waves).
// ---------------------------------------------------------------------------
__global__ void __launch_bounds__(256)
k_sims(const float* __restrict__ fe, int* __restrict__ maskg) {
    int gwave = blockIdx.x * 8 + (threadIdx.x >> 5);   // 0 .. B*32-1
    int b = gwave >> 5;
    int strip = gwave & 31;                            // 32 strips of 64 pairs
    int lane = threadIdx.x & 31;
    int row0 = strip * 64;
    const float4* feb = (const float4*)(fe + ((size_t)b * Kz + row0) * Dz); // 192 f4/row
    float4 prev[6], cur[6];
    #pragma unroll
    for (int j = 0; j < 6; ++j) prev[j] = feb[j * 32 + lane];
    for (int p = 0; p < 64; ++p) {
        int i = row0 + p;                // pair (i, i+1)
        if (i >= Kz - 1) break;
        const float4* rn = feb + (size_t)(p + 1) * (Dz / 4);
        float s = 0.f;
        #pragma unroll
        for (int j = 0; j < 6; ++j) {
            cur[j] = rn[j * 32 + lane];
            s += prev[j].x * cur[j].x + prev[j].y * cur[j].y
               + prev[j].z * cur[j].z + prev[j].w * cur[j].w;
        }
        #pragma unroll
        for (int off = 16; off > 0; off >>= 1) s += __shfl_xor(s, off, 32);
        if (lane == 0) maskg[b * Kz + i + 1] = (s < 0.7f) ? 1 : 0;   // 1 - THRESHOLD
        #pragma unroll
        for (int j = 0; j < 6; ++j) prev[j] = cur[j];
    }
}

// ---------------------------------------------------------------------------
// Per-batch scan: cummax(start)/suffix-min(end) -> (progress, dist) + abs idx.
// ---------------------------------------------------------------------------
__global__ void k_scan(const float* __restrict__ tp, const int* __restrict__ maskg,
                       float2* __restrict__ feats, int* __restrict__ idxv) {
    __shared__ int smask[Kz];
    __shared__ int sstart[Kz];
    __shared__ int send[Kz];
    int b = blockIdx.x, tid = threadIdx.x;
    for (int i = tid; i < Kz; i += blockDim.x) {
        int m = (i == 0) ? 1 : maskg[b * Kz + i];
        if (i == Kz - 1) m = 1;
        smask[i] = m;
    }
    __syncthreads();
    if (tid == 0) {
        int run = -1;
        for (int i = 0; i < Kz; ++i) { if (smask[i]) run = i; sstart[i] = run; }
        int minv = Kz;
        for (int i = Kz - 1; i >= 0; --i) {
            send[i] = (i == Kz - 1) ? (Kz - 1) : minv;
            int nb = smask[i] ? i : Kz;
            if (nb < minv) minv = nb;
        }
    }
    __syncthreads();
    for (int i = tid; i < Kz; i += blockDim.x) {
        int st = sstart[i], en = send[i];
        int len = en - st; if (len < 1) len = 1;
        float invl = 1.f / (float)len;
        float2 f; f.x = (float)(i - st) * invl; f.y = (float)(en - i) * invl;
        int row = b * Kz + i;
        feats[row] = f;
        int id = (int)(tp[row] * (float)(MAXLEN - 1));
        if (id < 0) id = 0;
        if (id > MAXLEN - 1) id = MAXLEN - 1;
        idxv[row] = id;
    }
}

// ---------------------------------------------------------------------------
// H[m,k] = gelu(feats[m] @ W1[k] + b1[k])  -> f16 (65536 x 512)
// ---------------------------------------------------------------------------
__global__ void k_h(const float2* __restrict__ feats, const float* __restrict__ W1,
                    const float* __restrict__ b1, _Float16* __restrict__ Hm) {
    int t = blockIdx.x * blockDim.x + threadIdx.x;
    if (t >= Mrows * HALFz) return;
    int row = t >> 9, k = t & (HALFz - 1);
    float2 f = feats[row];
    float x = f.x * W1[2 * k] + f.y * W1[2 * k + 1] + b1[k];
    Hm[t] = (_Float16)gelu_exact(x);
}

// ---------------------------------------------------------------------------
// Main GEMM: out[m,o] = sum_k H[m,k]*Wc[o,k] + Q[idx[m],o] + bc[o]
// Block 256 threads (8 waves). Tile M=64 x N=128. A tile (64x512 f16 = 64KB)
// staged in LDS once; each wave: 16 cols, 4 row-subtiles, 4 accumulators,
// B fragment reused across 4 WMMAs. grid = (1024, 8).
// ---------------------------------------------------------------------------
__global__ void __launch_bounds__(256)
k_main(const _Float16* __restrict__ Hm, const int* __restrict__ idxv,
       const _Float16* __restrict__ Wch, const float* __restrict__ Q,
       const float* __restrict__ bc, float* __restrict__ out) {
    __shared__ __align__(16) _Float16 Ah[64 * HALFz];   // 64 KB
    int mtile = blockIdx.x;     // rows mtile*64 ..
    int ntile = blockIdx.y;     // cols ntile*128 ..
    int tid = threadIdx.x;

    // Stage A tile: 64*512 halves = 4096 uint4
    {
        const uint4* src = (const uint4*)(Hm + (size_t)mtile * 64 * HALFz);
        uint4* dst = (uint4*)Ah;
        for (int t = tid; t < 64 * HALFz / 8; t += 256) dst[t] = src[t];
    }
    __syncthreads();

    int wave = tid >> 5, lane = tid & 31, nsub = lane & 15;
    int col0 = ntile * 128 + wave * 16;
    int akb = (lane < 16) ? 0 : 8;
    int bkb = (lane < 16) ? 0 : 8;
    int am  = lane & 15;
    const _Float16* brow = Wch + (size_t)(col0 + nsub) * HALFz;

    v8f cacc[4] = {};
    #pragma unroll 2
    for (int kk = 0; kk < HALFz / 32; ++kk) {
        const _Float16* bp = brow + kk * 32 + bkb;
        v8h blo = *(const v8h*)(bp);
        v8h bhi = *(const v8h*)(bp + 16);
        v16h bf = __builtin_shufflevector(blo, bhi,
                  0,1,2,3,4,5,6,7,8,9,10,11,12,13,14,15);
        #pragma unroll
        for (int s = 0; s < 4; ++s) {
            const _Float16* ap = &Ah[(s * 16 + am) * HALFz + kk * 32 + akb];
            v8h alo = *(const v8h*)(ap);
            v8h ahi = *(const v8h*)(ap + 16);
            v16h a = __builtin_shufflevector(alo, ahi,
                     0,1,2,3,4,5,6,7,8,9,10,11,12,13,14,15);
            cacc[s] = __builtin_amdgcn_wmma_f32_16x16x32_f16(
                          false, a, false, bf, (short)0, cacc[s], false, false);
        }
    }

    int mplus = (lane < 16) ? 0 : 8;
    int o = col0 + nsub;
    float bco = bc[o];
    #pragma unroll
    for (int s = 0; s < 4; ++s) {
        #pragma unroll
        for (int r = 0; r < 8; ++r) {
            int m = mtile * 64 + s * 16 + r + mplus;
            int id = idxv[m];
            out[(size_t)m * Hz + o] = cacc[s][r] + Q[(size_t)id * Hz + o] + bco;
        }
    }
}

// ---------------------------------------------------------------------------
extern "C" void kernel_launch(void* const* d_in, const int* in_sizes, int n_in,
                              void* d_out, int out_size, void* d_ws, size_t ws_size,
                              hipStream_t stream) {
    const float* tp = (const float*)d_in[0];   // (B,K)
    const float* fe = (const float*)d_in[1];   // (B,K,D)
    const float* pe = (const float*)d_in[2];   // (4500,512)
    const float* W1 = (const float*)d_in[3];   // (512,2)
    const float* b1 = (const float*)d_in[4];   // (512,)
    const float* W2 = (const float*)d_in[5];   // (512,512)
    const float* b2 = (const float*)d_in[6];   // (512,)
    const float* Wm = (const float*)d_in[7];   // (1024,1024)
    const float* bm = (const float*)d_in[8];   // (1024,)
    float* out = (float*)d_out;                // (B,K,1024) f32

    char* ws = (char*)d_ws;
    size_t off = 0;
    auto alloc = [&](size_t bytes) { char* p = ws + off; off = (off + bytes + 255) & ~(size_t)255; return p; };
    _Float16* Wch  = (_Float16*)alloc((size_t)Hz * HALFz * 2);        // 1 MB
    float*    Q    = (float*)   alloc((size_t)MPAD * Hz * 4);         // 18.5 MB
    float*    bcv  = (float*)   alloc(Hz * 4);
    float2*   fts  = (float2*)  alloc((size_t)Mrows * 8);             // 0.5 MB
    int*      idxv = (int*)     alloc((size_t)Mrows * 4);             // 0.25 MB
    int*      mskg = (int*)     alloc((size_t)Mrows * 4);             // 0.25 MB
    _Float16* WmAh = (_Float16*)alloc((size_t)Hz * HALFz * 2);        // 1 MB
    _Float16* WmBh = (_Float16*)alloc((size_t)Hz * HALFz * 2);        // 1 MB
    _Float16* W2t  = (_Float16*)alloc((size_t)HALFz * HALFz * 2);     // 0.5 MB
    _Float16* peh  = (_Float16*)alloc((size_t)MPAD * HALFz * 2);      // 4.6 MB
    _Float16* Hm   = (_Float16*)alloc((size_t)Mrows * HALFz * 2);     // 64 MB

    // Stage 1: conversions + bias fold (independent)
    k_cvt_wm<<<(Hz * Hz + 255) / 256, 256, 0, stream>>>(Wm, WmAh, WmBh);
    k_t_w2  <<<(HALFz * HALFz + 255) / 256, 256, 0, stream>>>(W2, W2t);
    k_cvt_pe<<<(MPAD * HALFz + 255) / 256, 256, 0, stream>>>(pe, peh);
    k_bconst<<<(Hz + 255) / 256, 256, 0, stream>>>(bm, b2, Wm, bcv);

    // Stage 2: folded-weight GEMMs on WMMA
    {   // Wc = WmB @ W2 (via W2t), f16 out: M=1024, N=512, K=512
        dim3 g(Hz / 16, HALFz / 64);
        k_gemm_nt<true><<<g, 128, 0, stream>>>(WmBh, W2t, (void*)Wch, HALFz, HALFz);
    }
    {   // Q = abs_pe @ WmA^T, f32 out: M=4512, N=1024, K=512
        dim3 g(MPAD / 16, Hz / 64);
        k_gemm_nt<false><<<g, 128, 0, stream>>>(peh, WmAh, (void*)Q, Hz, HALFz);
    }

    // Stage 3: boundary features
    k_sims<<<Bz * 4, 256, 0, stream>>>(fe, mskg);
    k_scan<<<Bz, 256, 0, stream>>>(tp, mskg, fts, idxv);
    k_h   <<<(Mrows * HALFz + 255) / 256, 256, 0, stream>>>(fts, W1, b1, Hm);

    // Stage 4: main WMMA GEMM + gather/bias epilogue
    dim3 g(Mrows / 64, Hz / 128);    // 1024 x 8
    k_main<<<g, 256, 0, stream>>>(Hm, idxv, Wch, Q, bcv, out);
}
